// HierCVQLayer_28046136443281
// MI455X (gfx1250) — matrix operand order repaired and checked by reading, not compiled, not run on hardware
//
#include <hip/hip_runtime.h>

// ---------------- problem constants ----------------
#define LOG2C   16
#define EMB     512
#define VQn     32
#define HIDn    1024
#define NHIDn   4
#define NTOK    32768            // B*L = 8*4096
#define HALF_TOK 16384
#define KCODES  1024             // 2^level, level=10
#define HALF_N  16777216u        // NTOK*KCODES/2 (threefry split point)
#define QUANT_ELEMS 16777216     // NTOK*EMB
#define CODE_OFF QUANT_ELEMS
#define LOSS_OFF (QUANT_ELEMS + NTOK)
#define BN_EPS   1e-5f
#define NORM_EPS 1e-6f

typedef _Float16 v16h __attribute__((ext_vector_type(16)));
typedef _Float16 h8   __attribute__((ext_vector_type(8)));
typedef float    v8f  __attribute__((ext_vector_type(8)));

static __device__ __forceinline__ v8f wmma_f16(v16h a, v16h b, v8f c) {
  // D = A(16x32 f16) x B(32x16 f16) + C(16x16 f32)
  return __builtin_amdgcn_wmma_f32_16x16x32_f16(false, a, false, b, (short)0, c, false, false);
}

// ---------------- threefry2x32 (exact JAX PRNG) ----------------
static __device__ __forceinline__ void threefry2x32(unsigned k0, unsigned k1,
                                                    unsigned x0, unsigned x1,
                                                    unsigned& o0, unsigned& o1) {
  const unsigned ks2 = k0 ^ k1 ^ 0x1BD11BDAu;
  x0 += k0; x1 += k1;
#define TFR(r) { x0 += x1; x1 = (x1 << (r)) | (x1 >> (32 - (r))); x1 ^= x0; }
  TFR(13) TFR(15) TFR(26) TFR(6)  x0 += k1;  x1 += ks2 + 1u;
  TFR(17) TFR(29) TFR(16) TFR(24) x0 += ks2; x1 += k0 + 2u;
  TFR(13) TFR(15) TFR(26) TFR(6)  x0 += k0;  x1 += k1 + 3u;
  TFR(17) TFR(29) TFR(16) TFR(24) x0 += k1;  x1 += ks2 + 4u;
  TFR(13) TFR(15) TFR(26) TFR(6)  x0 += ks2; x1 += k0 + 5u;
#undef TFR
  o0 = x0; o1 = x1;
}

static __device__ __forceinline__ float gumbel_bits(unsigned b) {
  // uniform in [0,1) as in jax: (bits>>9 | 0x3f800000) bitcast - 1, clamped to tiny
  const unsigned u = (b >> 9) | 0x3f800000u;
  float f = __uint_as_float(u) - 1.0f;
  f = fmaxf(f, 1.17549435e-38f);
  return -logf(-logf(f));
}

// ---------------- prep kernels ----------------
// dst[c*dstK + r] = (f16)src[r*C + c], zero-padded for r >= R
__global__ void k_transpose_f16(_Float16* __restrict__ dst, const float* __restrict__ src,
                                int R, int C, int dstK) {
  int i = blockIdx.x * blockDim.x + threadIdx.x;
  int total = C * dstK;
  if (i >= total) return;
  int c = i / dstK, r = i % dstK;
  float v = (r < R) ? src[(size_t)r * C + c] : 0.f;
  dst[i] = (_Float16)v;
}

// bv [KCODES, 32] f16, K columns padded to 32 (zeros) for one WMMA K-chunk
__global__ void k_build_bv(_Float16* __restrict__ dst, const int* __restrict__ level) {
  int i = blockIdx.x * blockDim.x + threadIdx.x;
  if (i >= KCODES * 32) return;
  int k = i >> 5, c = i & 31;
  int L = level[0];
  float v;
  if (c >= LOG2C) v = 0.f;
  else if (c < LOG2C - L) v = -1.f;
  else v = (float)((k >> (LOG2C - 1 - c)) & 1);
  dst[i] = (_Float16)v;
}

// ---------------- generic WMMA GEMM + bias [+ BN + ReLU] ----------------
// C[M,Nd] = act(A[M,Kd] @ W[Kd,Nd] + b); BT = W^T stored [Nd,Kd] f16
__global__ __launch_bounds__(32) void k_gemm16(
    const _Float16* __restrict__ A, const _Float16* __restrict__ BT,
    const float* __restrict__ bias,
    const float* __restrict__ g, const float* __restrict__ btp,
    const float* __restrict__ mu, const float* __restrict__ var,
    _Float16* outH, float* outF, int Nd, int Kd, int bnrelu) {
  const int lane = threadIdx.x & 31;
  const int hi = lane >> 4, r = lane & 15;
  const int mBase = blockIdx.y * 16, nBase = blockIdx.x * 16;
  const int rg = mBase + r;        // A-fragment row for this lane
  const int cg = nBase + r;        // B-fragment column for this lane
  v8f acc = {};
  for (int kc = 0; kc < Kd; kc += 32) {
    // A 16x32 f16 layout: halves 0..15 = K[kc+16*hi ..]
    h8 p = *(const h8*)(A + (size_t)rg * Kd + kc + 16 * hi);
    h8 q = *(const h8*)(A + (size_t)rg * Kd + kc + 16 * hi + 8);
    v16h a = __builtin_shufflevector(p, q, 0,1,2,3,4,5,6,7,8,9,10,11,12,13,14,15);
    // B 32x16: lane<16 -> K[kc..kc+15], lane>=16 -> K[kc+16..kc+31], col = lane&15
    v16h b = *(const v16h*)(BT + (size_t)cg * Kd + kc + hi * 16);
    acc = wmma_f16(a, b, acc);
  }
  const int ng = nBase + r;
  const float bb = bias[ng];
  float gg = 0.f, bt_ = 0.f, mm = 0.f, vv = 1.f;
  if (bnrelu) { gg = g[ng]; bt_ = btp[ng]; mm = mu[ng]; vv = var[ng]; }
#pragma unroll
  for (int v = 0; v < 8; ++v) {
    const int mg = mBase + v + 8 * hi;        // C/D: m = v + 8*hi, n = lane&15
    float x = acc[v] + bb;
    if (bnrelu) { x = gg * (x - mm) * rsqrtf(vv + BN_EPS) + bt_; x = fmaxf(x, 0.f); }
    if (outH) outH[(size_t)mg * Nd + ng] = (_Float16)x;
    if (outF) outF[(size_t)mg * Nd + ng] = x;
  }
}

// ---------------- embed normalization ----------------
__global__ void k_norm_embed(const float* __restrict__ eF, float* __restrict__ eN,
                             _Float16* __restrict__ eH) {
  int rr = blockIdx.x * blockDim.x + threadIdx.x;
  if (rr >= KCODES) return;
  float s = 0.f;
  for (int c = 0; c < VQn; ++c) { float x = eF[(size_t)rr * VQn + c]; s += x * x; }
  float inv = 1.f / (sqrtf(s) + NORM_EPS);
  for (int c = 0; c < VQn; ++c) {
    float v = eF[(size_t)rr * VQn + c] * inv;
    eN[(size_t)rr * VQn + c] = v;
    eH[(size_t)rr * VQn + c] = (_Float16)v;
  }
}

// ---------------- vq_loss: max off-diagonal similarity per row, then mean ----------------
__global__ void k_vq_rowmax(const float* __restrict__ eN, float* __restrict__ rowmax) {
  __shared__ float ei[VQn];
  __shared__ float red[64];
  const int i = blockIdx.x;
  if (threadIdx.x < VQn) ei[threadIdx.x] = eN[(size_t)i * VQn + threadIdx.x];
  __syncthreads();
  float best = -3.4e38f;
  for (int j = threadIdx.x; j < KCODES; j += 64) {
    float d;
    if (j == i) d = -1.f;
    else {
      float s = 0.f;
      for (int c = 0; c < VQn; ++c) s += ei[c] * eN[(size_t)j * VQn + c];
      d = s;
    }
    best = fmaxf(best, d);
  }
  red[threadIdx.x] = best;
  __syncthreads();
  for (int s = 32; s > 0; s >>= 1) {
    if (threadIdx.x < s) red[threadIdx.x] = fmaxf(red[threadIdx.x], red[threadIdx.x + s]);
    __syncthreads();
  }
  if (threadIdx.x == 0) rowmax[i] = red[0];
}

__global__ void k_vq_mean(const float* __restrict__ rowmax, float* __restrict__ outLoss) {
  __shared__ float red[256];
  float s = 0.f;
  for (int j = threadIdx.x; j < KCODES; j += 256) s += rowmax[j];
  red[threadIdx.x] = s;
  __syncthreads();
  for (int d = 128; d > 0; d >>= 1) {
    if (threadIdx.x < d) red[threadIdx.x] += red[threadIdx.x + d];
    __syncthreads();
  }
  if (threadIdx.x == 0) outLoss[0] = red[0] / (float)KCODES;
}

// ---------------- fused token kernel ----------------
// 256 threads (8 waves), 32 tokens/block: rows 0..15 = tokens [base..base+16),
// rows 16..31 = paired tokens [base+16384..), so one threefry call feeds both.
__global__ __launch_bounds__(256) void k_token(
    const float* __restrict__ hin, const int* __restrict__ amask,
    const _Float16* __restrict__ WpT, const float* __restrict__ bp,
    const _Float16* __restrict__ embedH, const float* __restrict__ embedN,
    const _Float16* __restrict__ WpiT, const float* __restrict__ bpi,
    float* __restrict__ outQ, float* __restrict__ outCode) {
  __shared__ float    lds_h[32][33];
  __shared__ float    lds_invn[32];
  __shared__ float    lds_bval[8][32];
  __shared__ int      lds_bidx[8][32];
  __shared__ _Float16 lds_q[32][32];

  const int tid  = threadIdx.x;
  const int w    = tid >> 5;
  const int lane = tid & 31;
  const int hi   = lane >> 4;
  const int r    = lane & 15;
  const int base = blockIdx.x * 16;   // low-token base in [0, 16384)

  // ---- Step 1: h = h_in @ Wp + bp for 32 tokens (waves 0..3, one 16x16 tile each) ----
  if (w < 4) {
    const int rt = w >> 1, ct = w & 1;
    const int tg = base + r + (rt ? HALF_TOK : 0);
    const float* rowp = hin + (size_t)tg * EMB;
    v8f acc = {};
    for (int kc = 0; kc < EMB; kc += 32) {
      if (kc + 32 < EMB) __builtin_prefetch(rowp + kc + 32, 0, 0);
      const float4* p1 = (const float4*)(rowp + kc + 16 * hi);
      const float4* p2 = (const float4*)(rowp + kc + 16 * hi + 8);
      float4 a0 = p1[0], a1 = p1[1], b0 = p2[0], b1 = p2[1];
      v16h a;
      a[0]=(_Float16)a0.x; a[1]=(_Float16)a0.y; a[2]=(_Float16)a0.z; a[3]=(_Float16)a0.w;
      a[4]=(_Float16)a1.x; a[5]=(_Float16)a1.y; a[6]=(_Float16)a1.z; a[7]=(_Float16)a1.w;
      a[8]=(_Float16)b0.x; a[9]=(_Float16)b0.y; a[10]=(_Float16)b0.z; a[11]=(_Float16)b0.w;
      a[12]=(_Float16)b1.x; a[13]=(_Float16)b1.y; a[14]=(_Float16)b1.z; a[15]=(_Float16)b1.w;
      v16h bfr = *(const v16h*)(WpT + (size_t)(ct * 16 + r) * EMB + kc + hi * 16);
      acc = wmma_f16(a, bfr, acc);
    }
    const int n = ct * 16 + r;
    const float bias = bp[n];
#pragma unroll
    for (int v = 0; v < 8; ++v) lds_h[rt * 16 + v + 8 * hi][n] = acc[v] + bias;
  }
  __syncthreads();

  // ---- Step 2: per-token inverse norms ----
  if (tid < 32) {
    float s = 0.f;
    for (int c = 0; c < VQn; ++c) { float x = lds_h[tid][c]; s += x * x; }
    lds_invn[tid] = 1.f / (sqrtf(s) + NORM_EPS);
  }
  __syncthreads();

  // ---- Step 3: streamed scores + gumbel + running argmax ----
  v16h aL, aH;
  {
    const float inL = lds_invn[r], inH = lds_invn[16 + r];
#pragma unroll
    for (int i = 0; i < 8; ++i) {
      aL[i]     = (_Float16)(lds_h[r][16 * hi + i] * inL);
      aL[8 + i] = (_Float16)(lds_h[r][16 * hi + 8 + i] * inL);
      aH[i]     = (_Float16)(lds_h[16 + r][16 * hi + i] * inH);
      aH[8 + i] = (_Float16)(lds_h[16 + r][16 * hi + 8 + i] * inH);
    }
  }
  float bv0[8], bv1[8];
  int   bi0[8], bi1[8];
#pragma unroll
  for (int v = 0; v < 8; ++v) { bv0[v] = bv1[v] = -3.4e38f; bi0[v] = bi1[v] = 0x7fffffff; }

  for (int nt = 0; nt < 8; ++nt) {         // wave w covers codes [w*128, w*128+128)
    const int cb = w * 128 + nt * 16;
    v16h bfr = *(const v16h*)(embedH + (size_t)(cb + r) * VQn + hi * 16);
    v8f d0 = {}, d1 = {};
    d0 = wmma_f16(aL, bfr, d0);
    d1 = wmma_f16(aH, bfr, d1);
    const int k = cb + r;                  // code index for this lane's column
#pragma unroll
    for (int v = 0; v < 8; ++v) {
      const int m = v + 8 * hi;
      const unsigned e0 = (unsigned)(base + m) * KCODES + (unsigned)k;
      unsigned o0, o1;
      threefry2x32(0u, 42u, e0, e0 + HALF_N, o0, o1);
      const float val0 = 2.f * d0[v] + gumbel_bits(o0);
      const float val1 = 2.f * d1[v] + gumbel_bits(o1);
      if (val0 > bv0[v]) { bv0[v] = val0; bi0[v] = k; }   // strict > keeps first index
      if (val1 > bv1[v]) { bv1[v] = val1; bi1[v] = k; }
    }
  }
  // reduce across the 16 lanes of each half-wave (same token, different code columns)
#pragma unroll
  for (int v = 0; v < 8; ++v) {
    float a0v = bv0[v]; int a0i = bi0[v];
    float a1v = bv1[v]; int a1i = bi1[v];
    for (int d = 1; d < 16; d <<= 1) {
      float t0 = __shfl_xor(a0v, d, 16); int u0 = __shfl_xor(a0i, d, 16);
      if (t0 > a0v || (t0 == a0v && u0 < a0i)) { a0v = t0; a0i = u0; }
      float t1 = __shfl_xor(a1v, d, 16); int u1 = __shfl_xor(a1i, d, 16);
      if (t1 > a1v || (t1 == a1v && u1 < a1i)) { a1v = t1; a1i = u1; }
    }
    if (r == 0) {
      const int m = v + 8 * hi;
      lds_bval[w][m]      = a0v; lds_bidx[w][m]      = a0i;
      lds_bval[w][16 + m] = a1v; lds_bidx[w][16 + m] = a1i;
    }
  }
  __syncthreads();

  // ---- Step 4: final argmax across waves (code ranges ascending in w), build q ----
  if (tid < 32) {
    float best = -3.4e38f; int bidx = 0x7fffffff;
    for (int ww = 0; ww < 8; ++ww) {
      float v = lds_bval[ww][tid]; int ii = lds_bidx[ww][tid];
      if (v > best || (v == best && ii < bidx)) { best = v; bidx = ii; }
    }
    const int tg = base + (tid & 15) + ((tid >= 16) ? HALF_TOK : 0);
    const int msk = (amask[tg] == 1);
    outCode[tg] = (float)(msk ? bidx : 0);
    const float inv = lds_invn[tid];
    for (int c = 0; c < VQn; ++c) {
      float hf = lds_h[tid][c] * inv;
      float e  = embedN[(size_t)bidx * VQn + c];
      float q  = (hf + e) - hf;            // straight-through arithmetic, matches ref fp
      lds_q[tid][c] = (_Float16)(msk ? q : 0.f);
    }
  }
  __syncthreads();

  // ---- Step 5: quantized = q @ Wpi + bpi  (2 row-tiles x 32 col-tiles, 8/wave) ----
  v16h aq0, aq1;
#pragma unroll
  for (int i = 0; i < 8; ++i) {
    aq0[i]     = lds_q[r][16 * hi + i];
    aq0[8 + i] = lds_q[r][16 * hi + 8 + i];
    aq1[i]     = lds_q[16 + r][16 * hi + i];
    aq1[8 + i] = lds_q[16 + r][16 * hi + 8 + i];
  }
  for (int jj = 0; jj < 8; ++jj) {
    const int job = w + 8 * jj;            // rt uniform per wave -> full EXEC at WMMA
    const int rt = job >> 5, nt = job & 31;
    const int cg = nt * 16 + r;
    v16h bfr = *(const v16h*)(WpiT + (size_t)cg * VQn + hi * 16);
    v8f d = {};
    d = wmma_f16(rt ? aq1 : aq0, bfr, d);
    const float bias = bpi[cg];
#pragma unroll
    for (int v = 0; v < 8; ++v) {
      const int m = v + 8 * hi;
      const int tg = base + m + (rt ? HALF_TOK : 0);
      outQ[(size_t)tg * EMB + cg] = d[v] + bias;
    }
  }
}

// ---------------- host launch ----------------
extern "C" void kernel_launch(void* const* d_in, const int* in_sizes, int n_in,
                              void* d_out, int out_size, void* d_ws, size_t ws_size,
                              hipStream_t stream) {
  (void)in_sizes; (void)n_in; (void)out_size; (void)ws_size;
  const float* h_in  = (const float*)d_in[0];
  const int*   amask = (const int*)d_in[1];
  const int*   level = (const int*)d_in[2];
  const float* Wp    = (const float*)d_in[3];
  const float* bp    = (const float*)d_in[4];
  const float* Wpi   = (const float*)d_in[5];
  const float* bpi   = (const float*)d_in[6];
  const float* W0    = (const float*)d_in[7];
  const float* b0    = (const float*)d_in[8];
  const float* Wh    = (const float*)d_in[9];
  const float* bh    = (const float*)d_in[10];
  const float* Wout  = (const float*)d_in[11];
  const float* bout  = (const float*)d_in[12];
  const float* bng   = (const float*)d_in[13];
  const float* bnb   = (const float*)d_in[14];
  const float* bnm   = (const float*)d_in[15];
  const float* bnv   = (const float*)d_in[16];

  char* ws = (char*)d_ws;
  size_t off = 0;
  auto alloc = [&](size_t bytes) -> char* {
    char* p = ws + off;
    off = (off + bytes + 255) & ~(size_t)255;
    return p;
  };
  _Float16* WpT    = (_Float16*)alloc((size_t)32 * 512 * 2);
  _Float16* W0T    = (_Float16*)alloc((size_t)1024 * 32 * 2);
  _Float16* WhT    = (_Float16*)alloc((size_t)4 * 1024 * 1024 * 2);
  _Float16* WoutT  = (_Float16*)alloc((size_t)32 * 1024 * 2);
  _Float16* WpiT   = (_Float16*)alloc((size_t)512 * 32 * 2);
  _Float16* bvH    = (_Float16*)alloc((size_t)KCODES * 32 * 2);
  _Float16* act0   = (_Float16*)alloc((size_t)KCODES * HIDn * 2);
  _Float16* act1   = (_Float16*)alloc((size_t)KCODES * HIDn * 2);
  float*    embedF = (float*)alloc((size_t)KCODES * VQn * 4);
  float*    embedN = (float*)alloc((size_t)KCODES * VQn * 4);
  _Float16* embedH = (_Float16*)alloc((size_t)KCODES * VQn * 2);
  float*    rowmax = (float*)alloc((size_t)KCODES * 4);

  auto tlaunch = [&](_Float16* dst, const float* src, int R, int C, int dstK) {
    int total = C * dstK;
    k_transpose_f16<<<(total + 255) / 256, 256, 0, stream>>>(dst, src, R, C, dstK);
  };
  tlaunch(WpT, Wp, 512, 32, 512);
  tlaunch(W0T, W0, 16, 1024, 32);                 // pad K 16 -> 32
  for (int i = 0; i < 4; ++i)
    tlaunch(WhT + (size_t)i * 1024 * 1024, Wh + (size_t)i * 1024 * 1024, 1024, 1024, 1024);
  tlaunch(WoutT, Wout, 1024, 32, 1024);
  tlaunch(WpiT, Wpi, 32, 512, 32);

  k_build_bv<<<(KCODES * 32 + 255) / 256, 256, 0, stream>>>(bvH, level);

  // embed MLP: Linear->BN->ReLU, 4x(Linear->BN->ReLU), Linear
  dim3 blk(32, 1, 1);
  k_gemm16<<<dim3(HIDn / 16, KCODES / 16), blk, 0, stream>>>(
      bvH, W0T, b0, bng, bnb, bnm, bnv, act0, nullptr, HIDn, 32, 1);
  _Float16* cur = act0; _Float16* nxt = act1;
  for (int i = 0; i < NHIDn; ++i) {
    k_gemm16<<<dim3(HIDn / 16, KCODES / 16), blk, 0, stream>>>(
        cur, WhT + (size_t)i * 1024 * 1024, bh + (size_t)i * HIDn,
        bng + (size_t)(i + 1) * HIDn, bnb + (size_t)(i + 1) * HIDn,
        bnm + (size_t)(i + 1) * HIDn, bnv + (size_t)(i + 1) * HIDn,
        nxt, nullptr, HIDn, HIDn, 1);
    _Float16* t = cur; cur = nxt; nxt = t;
  }
  k_gemm16<<<dim3(VQn / 16, KCODES / 16), blk, 0, stream>>>(
      cur, WoutT, bout, nullptr, nullptr, nullptr, nullptr,
      nullptr, embedF, VQn, HIDn, 0);

  k_norm_embed<<<(KCODES + 255) / 256, 256, 0, stream>>>(embedF, embedN, embedH);
  k_vq_rowmax<<<KCODES, 64, 0, stream>>>(embedN, rowmax);
  k_vq_mean<<<1, 256, 0, stream>>>(rowmax, (float*)d_out + LOSS_OFF);

  k_token<<<HALF_TOK / 16, 256, 0, stream>>>(
      h_in, amask, WpT, bp, embedH, embedN, WpiT, bpi,
      (float*)d_out, (float*)d_out + CODE_OFF);
}
